// CFIM_59674275610726
// MI455X (gfx1250) — compile-verified, matrix operand order checked by reading, not run on hardware
//
#include <hip/hip_runtime.h>
#include <hip/hip_bf16.h>

#define DIM   192
#define HID   96
#define CIN   384              // 2*DIM
#define CP    288              // 3*HID
#define BATCH 8
#define HH    160
#define WW    160
#define HWP   (HH * WW)        // 25600
#define LDS_KC 392             // 384 + 8 pad (16B-aligned ds_load_b128 rows)
#define LDS_KE 104             // 96 + 8 pad
#define NT    128              // pixel tile per block (2 x 16 cols per wave)

typedef __bf16 v16bf __attribute__((ext_vector_type(16)));
typedef float  v8f   __attribute__((ext_vector_type(8)));
typedef float  f4    __attribute__((ext_vector_type(4)));
typedef unsigned short us8 __attribute__((ext_vector_type(8)));

union BF16Frag {
    v16bf v;
    unsigned short u[16];
    us8 h[2];
};

static __device__ inline unsigned short f2b(float f) {
    unsigned u = __float_as_uint(f);
    u += 0x7FFFu + ((u >> 16) & 1u);          // round-to-nearest-even
    return (unsigned short)(u >> 16);
}
static __device__ inline float b2f(unsigned short h) {
    return __uint_as_float(((unsigned)h) << 16);
}

// LDS byte offset of a __shared__ object: generic LDS pointers carry the
// 32-bit LDS address in their low DWORD (aperture in the high bits).
static __device__ inline unsigned lds_off(const void* p) {
    return (unsigned)(uintptr_t)p;
}

// Async copy of `bytes` (multiple of 16) from contiguous global memory into
// contiguous LDS, spread across the whole block. ASYNCcnt-tracked.
static __device__ inline void async_copy_b128(unsigned lds_base, const void* gsrc,
                                              int bytes, int tid) {
    for (int off = tid * 16; off < bytes; off += 256 * 16) {
        unsigned       l = lds_base + (unsigned)off;
        const char*    g = (const char*)gsrc + off;
        asm volatile("global_load_async_to_lds_b128 %0, %1, off"
                     :: "v"(l), "v"(g) : "memory");
    }
    asm volatile("s_wait_asynccnt 0" ::: "memory");
}

// ---------------- Kernel A: per-(b,ch) mean & max over HW + bf16 copy of x ----------------
__global__ void k_reduce(const float* __restrict__ x1, const float* __restrict__ x2,
                         float* __restrict__ yin, unsigned short* __restrict__ xb) {
    int r  = blockIdx.x;                 // 0 .. B*CIN-1
    int b  = r / CIN, ch = r % CIN;
    const float* src = (ch < DIM) ? (x1 + ((size_t)b * DIM + ch) * HWP)
                                  : (x2 + ((size_t)b * DIM + (ch - DIM)) * HWP);
    unsigned short* dst = xb + ((size_t)b * CIN + ch) * HWP;
    float s = 0.f, m = -3.4e38f;
    for (int i = threadIdx.x * 4; i < HWP; i += 256 * 4) {
        f4 v = *(const f4*)(src + i);
        s += v[0] + v[1] + v[2] + v[3];
        m = fmaxf(m, fmaxf(fmaxf(v[0], v[1]), fmaxf(v[2], v[3])));
        unsigned p0 = (unsigned)f2b(v[0]) | ((unsigned)f2b(v[1]) << 16);
        unsigned p1 = (unsigned)f2b(v[2]) | ((unsigned)f2b(v[3]) << 16);
        uint2 pk; pk.x = p0; pk.y = p1;
        *(uint2*)(dst + i) = pk;
    }
    __shared__ float ss[256], sm[256];
    ss[threadIdx.x] = s; sm[threadIdx.x] = m;
    __syncthreads();
    for (int off = 128; off > 0; off >>= 1) {
        if (threadIdx.x < off) {
            ss[threadIdx.x] += ss[threadIdx.x + off];
            sm[threadIdx.x] = fmaxf(sm[threadIdx.x], sm[threadIdx.x + off]);
        }
        __syncthreads();
    }
    if (threadIdx.x == 0) {
        yin[(size_t)b * (2 * CIN) + ch]       = ss[0] * (1.0f / (float)HWP);
        yin[(size_t)b * (2 * CIN) + CIN + ch] = sm[0];
    }
}

// ---------------- Kernel W: weights -> bf16 (runs once, negligible) ----------------
__global__ void k_wcvt(const float* __restrict__ w_in, const float* __restrict__ w_out,
                       unsigned short* __restrict__ wi_b, unsigned short* __restrict__ wo_b) {
    int i = blockIdx.x * 256 + threadIdx.x;
    if (i < CP * CIN)       wi_b[i] = f2b(w_in[i]);
    if (i < 2 * DIM * HID)  wo_b[i] = f2b(w_out[i]);
}

// ---------------- Kernels B: tiny MLP (8x768 rows, negligible) ----------------
__global__ void k_mlp1(const float* __restrict__ yin, const float* __restrict__ w1,
                       const float* __restrict__ b1, float* __restrict__ y1) {
    int idx = blockIdx.x * 256 + threadIdx.x;
    if (idx >= BATCH * 768) return;
    int b = idx / 768, o = idx % 768;
    const float* xr = yin + (size_t)b * 768;
    const float* wr = w1 + (size_t)o * 768;
    float acc = b1[o];
    for (int k = 0; k < 768; ++k) acc += xr[k] * wr[k];
    y1[idx] = fmaxf(acc, 0.f);
}

__global__ void k_mlp2(const float* __restrict__ y1, const float* __restrict__ w2,
                       const float* __restrict__ b2, float* __restrict__ cw) {
    int idx = blockIdx.x * 256 + threadIdx.x;
    if (idx >= BATCH * CIN) return;
    int b = idx / CIN, o = idx % CIN;
    const float* xr = y1 + (size_t)b * 768;
    const float* wr = w2 + (size_t)o * 768;
    float acc = b2[o];
    for (int k = 0; k < 768; ++k) acc += xr[k] * wr[k];
    cw[idx] = 1.f / (1.f + __expf(-acc));
}

// ---------------- Kernel C: p = W_in(288x384) x X(384x25600)  (bf16 WMMA) ----------------
__global__ void __launch_bounds__(256)
k_pw_in(const unsigned short* __restrict__ xb, const unsigned short* __restrict__ wi_b,
        unsigned short* __restrict__ p_ws) {
    __shared__ unsigned short ldsX[NT * LDS_KC];   // X tile, [n][k] transposed, ~98 KB
    __shared__ unsigned short ldsW[32 * CIN];      // 32-row weight band, [m][k], 24 KB
    int b    = blockIdx.y;
    int pix0 = blockIdx.x * NT;
    int tid  = threadIdx.x;

    // Stage X tile (K=384 x N=128): b128 global reads, transposed LDS stores.
    for (int c = tid; c < (CIN * NT) / 8; c += 256) {
        int k  = c >> 4;                 // NT/8 = 16 chunks per k-row
        int n0 = (c & 15) * 8;
        us8 v = *(const us8*)(xb + ((size_t)b * CIN + k) * HWP + pix0 + n0);
#pragma unroll
        for (int j = 0; j < 8; ++j)
            ldsX[(n0 + j) * LDS_KC + k] = v[j];
    }
    __syncthreads();

    int wave = tid >> 5, lane = tid & 31;          // wave32
    int wm = wave >> 2, wn = wave & 3;             // 2(M) x 4(N=32col) waves
    int lh = lane >> 4, l15 = lane & 15;
    unsigned wbase = lds_off(&ldsW[0]);

    for (int mi = 0; mi < CP / 32; ++mi) {         // 9 M-iters (32 rows each)
        __syncthreads();                           // previous band readers done
        async_copy_b128(wbase, wi_b + (size_t)mi * 32 * CIN, 32 * CIN * 2, tid);
        __syncthreads();                           // band visible to all waves

        int m0 = mi * 32 + wm * 16;
        v8f acc0 = {}, acc1 = {};
#pragma unroll 4
        for (int ks = 0; ks < CIN; ks += 32) {     // 12 K-steps
            int kb = ks + (lh << 3);
            BF16Frag A;                            // A reused across 2 N-subtiles
            const unsigned short* apw = &ldsW[(wm * 16 + l15) * CIN + kb];
            A.h[0] = *(const us8*)(apw);
            A.h[1] = *(const us8*)(apw + 16);
            BF16Frag B0, B1;
            const unsigned short* bp0 = &ldsX[(wn * 32 + l15) * LDS_KC + kb];
            const unsigned short* bp1 = &ldsX[(wn * 32 + 16 + l15) * LDS_KC + kb];
            B0.h[0] = *(const us8*)(bp0);
            B0.h[1] = *(const us8*)(bp0 + 16);
            B1.h[0] = *(const us8*)(bp1);
            B1.h[1] = *(const us8*)(bp1 + 16);
            acc0 = __builtin_amdgcn_wmma_f32_16x16x32_bf16(
                false, A.v, false, B0.v, (short)0, acc0, false, false);
            acc1 = __builtin_amdgcn_wmma_f32_16x16x32_bf16(
                false, A.v, false, B1.v, (short)0, acc1, false, false);
        }
        int na = pix0 + wn * 32 + l15;
        int mb = m0 + lh * 8;
#pragma unroll
        for (int r = 0; r < 8; ++r) {
            size_t row = ((size_t)b * CP + mb + r) * HWP;
            p_ws[row + na]      = f2b(acc0[r]);
            p_ws[row + na + 16] = f2b(acc1[r]);
        }
    }
}

// ---------------- Kernel D: dilated depthwise convs + exact-GELU gate ----------------
__global__ void k_dwgelu(const unsigned short* __restrict__ p_ws,
                         const float* __restrict__ wd1, const float* __restrict__ wd2,
                         const float* __restrict__ wd3, unsigned short* __restrict__ g_ws) {
    size_t idx = (size_t)blockIdx.x * 256 + threadIdx.x;
    if (idx >= (size_t)BATCH * HID * HWP) return;
    int pix = (int)(idx % HWP);
    int t   = (int)(idx / HWP);
    int c = t % HID, b = t / HID;
    int h = pix / WW, w = pix % WW;
    const unsigned short* pa = p_ws + ((size_t)b * CP + c) * HWP;
    const unsigned short* p2 = p_ws + ((size_t)b * CP + HID + c) * HWP;
    const unsigned short* p3 = p_ws + ((size_t)b * CP + 2 * HID + c) * HWP;
    float a = 0.f, g2 = 0.f, g3 = 0.f;
#pragma unroll
    for (int dy = -1; dy <= 1; ++dy) {
#pragma unroll
        for (int dx = -1; dx <= 1; ++dx) {
            int wi = (dy + 1) * 3 + (dx + 1);
            int h1 = h + dy,     w1 = w + dx;
            int h2 = h + 2 * dy, w2 = w + 2 * dx;
            int h3 = h + 3 * dy, w3 = w + 3 * dx;
            if (h1 >= 0 && h1 < HH && w1 >= 0 && w1 < WW)
                a  += b2f(pa[h1 * WW + w1]) * wd1[c * 9 + wi];
            if (h2 >= 0 && h2 < HH && w2 >= 0 && w2 < WW)
                g2 += b2f(p2[h2 * WW + w2]) * wd2[c * 9 + wi];
            if (h3 >= 0 && h3 < HH && w3 >= 0 && w3 < WW)
                g3 += b2f(p3[h3 * WW + w3]) * wd3[c * 9 + wi];
        }
    }
    float ge = 0.5f * a * (1.0f + erff(a * 0.70710678118654752f));  // exact GELU
    g_ws[((size_t)b * HID + c) * HWP + pix] = f2b(ge * (g2 + g3));
}

// ---------------- Kernel E: sw = W_out(384x96) x g, fused residual epilogue ----------------
static __device__ inline void epi_store(const v8f& acc, int b, int mb, int n,
                                        const float* __restrict__ cw,
                                        const float* __restrict__ x1,
                                        const float* __restrict__ x2,
                                        float* __restrict__ out, size_t outhalf) {
#pragma unroll
    for (int r = 0; r < 8; ++r) {
        int   m  = mb + r;
        float sw = 0.5f * acc[r];                  // LAMBDA_S = 0.5
        if (m < DIM) {                             // sw0 row -> out_x2 channel m
            size_t off = ((size_t)b * DIM + m) * HWP + n;
            out[outhalf + off] = x2[off] + 0.5f * cw[b * 2 * DIM + m] * x1[off] + sw;
        } else {                                   // sw1 row -> out_x1 channel m-DIM
            int c = m - DIM;
            size_t off = ((size_t)b * DIM + c) * HWP + n;
            out[off] = x1[off] + 0.5f * cw[b * 2 * DIM + DIM + c] * x2[off] + sw;
        }
    }
}

__global__ void __launch_bounds__(256)
k_pw_out(const unsigned short* __restrict__ g_ws, const unsigned short* __restrict__ wo_b,
         const float* __restrict__ cw, const float* __restrict__ x1,
         const float* __restrict__ x2, float* __restrict__ out) {
    __shared__ unsigned short ldsG[NT * LDS_KE];   // g tile, [n][k], 26.6 KB
    __shared__ unsigned short ldsW[32 * HID];      // 32-row weight band, 6 KB
    int b    = blockIdx.y;
    int pix0 = blockIdx.x * NT;
    int tid  = threadIdx.x;

    for (int c = tid; c < (HID * NT) / 8; c += 256) {
        int k  = c >> 4;
        int n0 = (c & 15) * 8;
        us8 v = *(const us8*)(g_ws + ((size_t)b * HID + k) * HWP + pix0 + n0);
#pragma unroll
        for (int j = 0; j < 8; ++j)
            ldsG[(n0 + j) * LDS_KE + k] = v[j];
    }
    __syncthreads();

    int wave = tid >> 5, lane = tid & 31;
    int wm = wave >> 2, wn = wave & 3;
    int lh = lane >> 4, l15 = lane & 15;
    unsigned wbase = lds_off(&ldsW[0]);
    const size_t OUTHALF = (size_t)BATCH * DIM * HWP;

    for (int mi = 0; mi < (2 * DIM) / 32; ++mi) {  // 12 M-iters
        __syncthreads();
        async_copy_b128(wbase, wo_b + (size_t)mi * 32 * HID, 32 * HID * 2, tid);
        __syncthreads();

        int m0 = mi * 32 + wm * 16;
        v8f acc0 = {}, acc1 = {};
#pragma unroll
        for (int ks = 0; ks < HID; ks += 32) {     // 3 K-steps
            int kb = ks + (lh << 3);
            BF16Frag A;
            const unsigned short* apw = &ldsW[(wm * 16 + l15) * HID + kb];
            A.h[0] = *(const us8*)(apw);
            A.h[1] = *(const us8*)(apw + 16);
            BF16Frag B0, B1;
            const unsigned short* bp0 = &ldsG[(wn * 32 + l15) * LDS_KE + kb];
            const unsigned short* bp1 = &ldsG[(wn * 32 + 16 + l15) * LDS_KE + kb];
            B0.h[0] = *(const us8*)(bp0);
            B0.h[1] = *(const us8*)(bp0 + 16);
            B1.h[0] = *(const us8*)(bp1);
            B1.h[1] = *(const us8*)(bp1 + 16);
            acc0 = __builtin_amdgcn_wmma_f32_16x16x32_bf16(
                false, A.v, false, B0.v, (short)0, acc0, false, false);
            acc1 = __builtin_amdgcn_wmma_f32_16x16x32_bf16(
                false, A.v, false, B1.v, (short)0, acc1, false, false);
        }
        int na = pix0 + wn * 32 + l15;
        int mb = m0 + lh * 8;
        epi_store(acc0, b, mb, na,      cw, x1, x2, out, OUTHALF);
        epi_store(acc1, b, mb, na + 16, cw, x1, x2, out, OUTHALF);
    }
}

extern "C" void kernel_launch(void* const* d_in, const int* in_sizes, int n_in,
                              void* d_out, int out_size, void* d_ws, size_t ws_size,
                              hipStream_t stream) {
    const float* x1     = (const float*)d_in[0];
    const float* x2     = (const float*)d_in[1];
    const float* mlp_w1 = (const float*)d_in[2];
    const float* mlp_b1 = (const float*)d_in[3];
    const float* mlp_w2 = (const float*)d_in[4];
    const float* mlp_b2 = (const float*)d_in[5];
    const float* w_in   = (const float*)d_in[6];
    const float* w_dw1  = (const float*)d_in[7];
    const float* w_dw2  = (const float*)d_in[8];
    const float* w_dw3  = (const float*)d_in[9];
    const float* w_out  = (const float*)d_in[10];
    float* out = (float*)d_out;

    // d_ws: small buffers + bf16 weights + bf16 g (~40 MB total).
    char*  ws  = (char*)d_ws;
    float* yin = (float*)ws;                                   // 8*768 f32
    float* y1  = yin + BATCH * 768;                            // 8*768 f32
    float* cw  = y1 + BATCH * 768;                             // 8*384 f32
    unsigned short* wi_b = (unsigned short*)(ws + 65536);      // 288*384 bf16
    unsigned short* wo_b = wi_b + CP * CIN;                    // 384*96 bf16
    unsigned short* g_ws = (unsigned short*)(ws + 393216);     // 8*96*25600 bf16

    // d_out doubles as scratch until k_pw_out fully overwrites it:
    //   [0, 118MB):    p (bf16)
    //   [118, 275MB):  x (bf16, concat of x1/x2)
    unsigned short* p_ws = (unsigned short*)d_out;
    unsigned short* xb   = (unsigned short*)((char*)d_out + (size_t)CP * BATCH * HWP * 2);

    k_reduce<<<dim3(BATCH * CIN), dim3(256), 0, stream>>>(x1, x2, yin, xb);
    k_wcvt<<<dim3((CP * CIN + 255) / 256), dim3(256), 0, stream>>>(w_in, w_out, wi_b, wo_b);
    k_mlp1<<<dim3((BATCH * 768 + 255) / 256), dim3(256), 0, stream>>>(yin, mlp_w1, mlp_b1, y1);
    k_mlp2<<<dim3((BATCH * CIN + 255) / 256), dim3(256), 0, stream>>>(y1, mlp_w2, mlp_b2, cw);
    k_pw_in<<<dim3(HWP / NT, BATCH), dim3(256), 0, stream>>>(xb, wi_b, p_ws);
    k_dwgelu<<<dim3((int)(((size_t)BATCH * HID * HWP + 255) / 256)), dim3(256), 0, stream>>>(
        p_ws, w_dw1, w_dw2, w_dw3, g_ws);
    k_pw_out<<<dim3(HWP / NT, BATCH), dim3(256), 0, stream>>>(g_ws, wo_b, cw, x1, x2, out);
}
